// DecoNet_88201448390854
// MI455X (gfx1250) — matrix operand order repaired and checked by reading, not compiled, not run on hardware
//
#include <hip/hip_runtime.h>
#include <hip/hip_bf16.h>

// ---------------------------------------------------------------------------
// Problem constants (match reference)
// ---------------------------------------------------------------------------
#define NNODE 50000
#define HDIM  256
#define EMB   10
#define NB    4
#define RREP  2
#define DOUT  10
#define E1C   200000
#define E2C   250000
#define E4C   300000
#define EAC   300000
#define DIN   64

typedef long long ll;
typedef __attribute__((ext_vector_type(16))) __bf16 v16bf;
typedef __attribute__((ext_vector_type(8)))  float  v8f;

// fp32 -> bf16 round-to-nearest-even
__device__ __forceinline__ unsigned int f2bf(float f) {
  unsigned int u = __float_as_uint(f);
  unsigned int r = u + 0x7FFFu + ((u >> 16) & 1u);
  return r >> 16;
}
// pack two fp32 -> packed bf16x2 (lo in [15:0], hi in [31:16])
__device__ __forceinline__ unsigned int pk2bf(float lo, float hi) {
  return (f2bf(hi) << 16) | f2bf(lo);
}

// ---------------------------------------------------------------------------
// WMMA GEMM: C[M,Ncol] = epilogue(A[M,K] @ B[K,Ncol] + bias, res)
//   MODE 0: acc + bias
//   MODE 1: res + relu(acc + bias)        (conv block)
//   MODE 2: relu(acc + bias)              (ffn1)
//   MODE 3: res + acc + bias              (attention out / ffn2)
//
// Block: 128 threads (4 waves). Tile 128(M) x 64(N), K-step 32.
// Each wave owns a 32x64 strip: 2 A-frags x 4 B-frags = 8 WMMAs / K-step.
// LDS tiles are stored in *fragment order* (bf16 pairs packed in u32), so a
// lane's whole 16-element fragment is one 32B contiguous chunk read with two
// ds_load_b128. Interior blocks take unguarded (no exec-mask churn) staging
// and store paths; guards only run on M/N/K edge tiles.
// ---------------------------------------------------------------------------
#define BM 128
#define BN 64
#define BK 32

template <int MODE>
__global__ __launch_bounds__(128)
void wmma_gemm(const float* __restrict__ A, const float* __restrict__ B,
               const float* __restrict__ bias, const float* __restrict__ res,
               float* __restrict__ C, int M, int Kdim, int Ncol)
{
  // A fragment store: [row][half][j]  (j = packed bf16 pair index 0..7)
  __shared__ unsigned int AsF[BM][2][8];        // 8 KB
  // B fragment store: [group][col16][half][j]
  __shared__ unsigned int BsF[4][16][2][8];     // 4 KB

  const int t    = threadIdx.x;     // 0..127
  const int wave = t >> 5;          // 0..3
  const int lane = t & 31;
  const int half = lane >> 4;       // 0 / 1
  const int l16  = lane & 15;

  const int m0 = blockIdx.y * BM;
  const int n0 = blockIdx.x * BN;

  const bool fullM = (m0 + BM <= M);
  const bool fullN = (n0 + BN <= Ncol);
  const bool evenK = ((Kdim & 1) == 0);

  v8f acc[2][4] = {};

  union Frag { v16bf v; uint4 q[2]; };

  for (int k0 = 0; k0 < Kdim; k0 += BK) {
    const bool fullK = (k0 + BK <= Kdim);

    // --- stage A tile [128 x 32]: 2048 bf16-pairs, 16 per thread.
    // Inverse fragment map for column c: half=(c>>3)&1, j=((c&7)>>1)+(c&16?4:0)
    if (fullM && fullK && evenK) {
      #pragma unroll
      for (int p = 0; p < 16; ++p) {
        int i = t + 128 * p;
        int r = i >> 4;               // 0..127
        int c = (i & 15) * 2;         // even column 0..30
        float2 v = *(const float2*)&A[(ll)(m0 + r) * Kdim + (k0 + c)];
        int hsel = (c >> 3) & 1;
        int j    = ((c & 7) >> 1) + ((c & 16) ? 4 : 0);
        AsF[r][hsel][j] = pk2bf(v.x, v.y);
      }
    } else {
      #pragma unroll
      for (int p = 0; p < 16; ++p) {
        int i  = t + 128 * p;
        int r  = i >> 4;
        int c  = (i & 15) * 2;
        int gr = m0 + r, gc = k0 + c;
        float lo = 0.f, hi = 0.f;
        if (gr < M && gc < Kdim)     lo = A[(ll)gr * Kdim + gc];
        if (gr < M && gc + 1 < Kdim) hi = A[(ll)gr * Kdim + gc + 1];
        int hsel = (c >> 3) & 1;
        int j    = ((c & 7) >> 1) + ((c & 16) ? 4 : 0);
        AsF[r][hsel][j] = pk2bf(lo, hi);
      }
    }
    // --- stage B tile [32 x 64]: 1024 bf16-pairs (pair across K rows), 8/thread.
    // row kr: half=(kr>>3)&1, j=((kr&7)>>1)+(kr&16?4:0) ; col c: group=c>>4, col16=c&15
    if (fullN && fullK) {
      #pragma unroll
      for (int p = 0; p < 8; ++p) {
        int i   = t + 128 * p;        // 0..1023
        int c   = i & 63;
        int krp = i >> 6;             // K pair index 0..15
        int kr  = k0 + krp * 2;
        float lo = B[(ll)kr * Ncol + (n0 + c)];
        float hi = B[(ll)(kr + 1) * Ncol + (n0 + c)];
        BsF[c >> 4][c & 15][(krp >> 2) & 1][(krp & 3) + ((krp & 8) ? 4 : 0)] = pk2bf(lo, hi);
      }
    } else {
      #pragma unroll
      for (int p = 0; p < 8; ++p) {
        int i   = t + 128 * p;
        int c   = i & 63;
        int krp = i >> 6;
        int kr  = krp * 2;
        int gc  = n0 + c;
        float lo = 0.f, hi = 0.f;
        if (gc < Ncol && k0 + kr < Kdim)     lo = B[(ll)(k0 + kr) * Ncol + gc];
        if (gc < Ncol && k0 + kr + 1 < Kdim) hi = B[(ll)(k0 + kr + 1) * Ncol + gc];
        BsF[c >> 4][c & 15][(krp >> 2) & 1][(krp & 3) + ((krp & 8) ? 4 : 0)] = pk2bf(lo, hi);
      }
    }
    __syncthreads();

    // CDNA5 global_prefetch_b8 of next K tiles while WMMAs run
    if (k0 + BK < Kdim) {
      int ar = m0 + t;
      if (ar < M) __builtin_prefetch(&A[(ll)ar * Kdim + k0 + BK], 0, 1);
      int br = k0 + BK + (t & 31);
      if (br < Kdim) __builtin_prefetch(&B[(ll)br * Ncol + n0], 0, 1);
    }

    // --- load fragments (two ds_load_b128 each) + 8 WMMAs
    Frag a0, a1;
    {
      const uint4* pa0 = (const uint4*)&AsF[wave * 32 + l16][half][0];
      a0.q[0] = pa0[0]; a0.q[1] = pa0[1];
      const uint4* pa1 = (const uint4*)&AsF[wave * 32 + 16 + l16][half][0];
      a1.q[0] = pa1[0]; a1.q[1] = pa1[1];
    }
    #pragma unroll
    for (int g = 0; g < 4; ++g) {
      Frag bf_;
      const uint4* pb = (const uint4*)&BsF[g][l16][half][0];
      bf_.q[0] = pb[0]; bf_.q[1] = pb[1];
      acc[0][g] = __builtin_amdgcn_wmma_f32_16x16x32_bf16(
          false, a0.v, false, bf_.v, (short)0, acc[0][g], false, false);
      acc[1][g] = __builtin_amdgcn_wmma_f32_16x16x32_bf16(
          false, a1.v, false, bf_.v, (short)0, acc[1][g], false, false);
    }
    __syncthreads();
  }

  // --- fused epilogue + store.  C/D layout: VGPR i -> (M = i + 8*half, N = lane&15)
  if (fullM && fullN) {
    #pragma unroll
    for (int s = 0; s < 2; ++s) {
      #pragma unroll
      for (int g = 0; g < 4; ++g) {
        int col = n0 + g * 16 + l16;
        float bv = bias ? bias[col] : 0.f;
        #pragma unroll
        for (int i = 0; i < 8; ++i) {
          int row = m0 + wave * 32 + s * 16 + (half ? i + 8 : i);
          ll o = (ll)row * Ncol + col;
          float v = acc[s][g][i] + bv;
          if (MODE == 1)      v = res[o] + fmaxf(v, 0.f);
          else if (MODE == 2) v = fmaxf(v, 0.f);
          else if (MODE == 3) v = res[o] + v;
          C[o] = v;
        }
      }
    }
  } else {
    #pragma unroll
    for (int s = 0; s < 2; ++s) {
      #pragma unroll
      for (int g = 0; g < 4; ++g) {
        int col = n0 + g * 16 + l16;
        if (col >= Ncol) continue;
        float bv = bias ? bias[col] : 0.f;
        #pragma unroll
        for (int i = 0; i < 8; ++i) {
          int row = m0 + wave * 32 + s * 16 + (half ? i + 8 : i);
          if (row >= M) continue;
          ll o = (ll)row * Ncol + col;
          float v = acc[s][g][i] + bv;
          if (MODE == 1)      v = res[o] + fmaxf(v, 0.f);
          else if (MODE == 2) v = fmaxf(v, 0.f);
          else if (MODE == 3) v = res[o] + v;
          C[o] = v;
        }
      }
    }
  }
}

// ---------------------------------------------------------------------------
// Elementwise / sparse kernels
// ---------------------------------------------------------------------------
__global__ void fill_kernel(float* p, ll n, float v) {
  ll i = (ll)blockIdx.x * blockDim.x + threadIdx.x;
  if (i < n) p[i] = v;
}

// out[dst,:] += val[e] * h[src,:]   (idx = [dst row ; src row], H=256)
__global__ void spmm_scatter(const float* __restrict__ h, const int* __restrict__ idx,
                             const float* __restrict__ val, int vstride,
                             float* __restrict__ out, int E) {
  ll i = (ll)blockIdx.x * blockDim.x + threadIdx.x;
  if (i >= (ll)E * HDIM) return;
  int e = (int)(i >> 8);
  int c = (int)(i & 255);
  int dst = idx[e], src = idx[E + e];
  atomicAdd(&out[(ll)dst * HDIM + c],
            val[(ll)e * vstride] * h[(ll)src * HDIM + c]);
}

// dst[r, col_off + c] = src[r, c]   (JK concat)
__global__ void copy_part(const float* __restrict__ src, float* __restrict__ dst,
                          int rows, int dst_stride, int col_off) {
  ll i = (ll)blockIdx.x * blockDim.x + threadIdx.x;
  if (i >= (ll)rows * HDIM) return;
  int r = (int)(i >> 8), c = (int)(i & 255);
  dst[(ll)r * dst_stride + col_off + c] = src[i];
}

__global__ void add_inplace(float* a, const float* __restrict__ b, ll n) {
  ll i = (ll)blockIdx.x * blockDim.x + threadIdx.x;
  if (i < n) a[i] += b[i];
}

// per-column mean / rsqrt(var+eps) over rows
__global__ __launch_bounds__(256)
void bn_stats(const float* __restrict__ y, float* __restrict__ mean,
              float* __restrict__ rstd, int rows) {
  __shared__ float ssum[256], ssq[256];
  int c = blockIdx.x, t = threadIdx.x;
  float s = 0.f, q = 0.f;
  for (int r = t; r < rows; r += 256) {
    float v = y[(ll)r * HDIM + c];
    s += v; q += v * v;
  }
  ssum[t] = s; ssq[t] = q;
  __syncthreads();
  for (int off = 128; off; off >>= 1) {
    if (t < off) { ssum[t] += ssum[t + off]; ssq[t] += ssq[t + off]; }
    __syncthreads();
  }
  if (t == 0) {
    float m = ssum[0] / rows;
    float v = ssq[0] / rows - m * m;
    mean[c] = m;
    rstd[c] = rsqrtf(v + 1e-5f);
  }
}

__global__ void bn_relu(const float* __restrict__ y, const float* __restrict__ mean,
                        const float* __restrict__ rstd, const float* __restrict__ g,
                        const float* __restrict__ b, float* __restrict__ out, ll n) {
  ll i = (ll)blockIdx.x * blockDim.x + threadIdx.x;
  if (i >= n) return;
  int c = (int)(i & 255);
  float v = (y[i] - mean[c]) * rstd[c] * g[c] + b[c];
  out[i] = fmaxf(v, 0.f);
}

// score[e] = dot(q[dst],k[src])/16 + dot(full_val[e], we); segment-max into mbuf.
// One wave32 per edge; shuffle tree reduction.
__global__ __launch_bounds__(256)
void attn_score(const float* __restrict__ q, const float* __restrict__ k,
                const float* __restrict__ fv, const float* __restrict__ we,
                const int* __restrict__ idx, float* __restrict__ score,
                float* __restrict__ mbuf, int E) {
  int e = blockIdx.x * 8 + (threadIdx.x >> 5);
  int lane = threadIdx.x & 31;
  if (e >= E) return;
  int dst = idx[e], src = idx[E + e];
  float s1 = 0.f, s2 = 0.f;
  for (int c = lane; c < HDIM; c += 32) {
    s1 += q[(ll)dst * HDIM + c] * k[(ll)src * HDIM + c];
    s2 += fv[(ll)e * HDIM + c] * we[c];
  }
  for (int off = 16; off; off >>= 1) {
    s1 += __shfl_xor(s1, off, 32);
    s2 += __shfl_xor(s2, off, 32);
  }
  if (lane == 0) {
    float sc = s1 * 0.0625f + s2;
    score[e] = sc;
    int* addr = (int*)&mbuf[dst];
    int old = __float_as_int(mbuf[dst]);
    while (__int_as_float(old) < sc) {
      int prev = atomicCAS(addr, old, __float_as_int(sc));
      if (prev == old) break;
      old = prev;
    }
  }
}

__global__ void attn_pr(const float* __restrict__ score, const float* __restrict__ mbuf,
                        const int* __restrict__ idx, float* __restrict__ pr,
                        float* __restrict__ Z, int E) {
  int e = blockIdx.x * blockDim.x + threadIdx.x;
  if (e >= E) return;
  int dst = idx[e];
  float p = __expf(score[e] - mbuf[dst]);
  pr[e] = p;
  atomicAdd(&Z[dst], p);
}

__global__ void attn_aggregate(const float* __restrict__ pr, const float* __restrict__ Z,
                               const float* __restrict__ v, const int* __restrict__ idx,
                               float* __restrict__ out, int E) {
  ll i = (ll)blockIdx.x * blockDim.x + threadIdx.x;
  if (i >= (ll)E * HDIM) return;
  int e = (int)(i >> 8);
  int c = (int)(i & 255);
  int dst = idx[e], src = idx[E + e];
  float a = pr[e] / (Z[dst] + 1e-9f);
  atomicAdd(&out[(ll)dst * HDIM + c], a * v[(ll)src * HDIM + c]);
}

// ---------------------------------------------------------------------------
// Orchestration
// ---------------------------------------------------------------------------
static inline void launch_gemm(const float* A, const float* B, const float* bias,
                               const float* res, float* C, int M, int Kd, int Nc,
                               int mode, hipStream_t stream) {
  dim3 grid((Nc + BN - 1) / BN, (M + BM - 1) / BM);
  dim3 blk(128);
  switch (mode) {
    case 0: wmma_gemm<0><<<grid, blk, 0, stream>>>(A, B, bias, res, C, M, Kd, Nc); break;
    case 1: wmma_gemm<1><<<grid, blk, 0, stream>>>(A, B, bias, res, C, M, Kd, Nc); break;
    case 2: wmma_gemm<2><<<grid, blk, 0, stream>>>(A, B, bias, res, C, M, Kd, Nc); break;
    default: wmma_gemm<3><<<grid, blk, 0, stream>>>(A, B, bias, res, C, M, Kd, Nc); break;
  }
}

static inline void launch_fill(float* p, ll n, float v, hipStream_t stream) {
  fill_kernel<<<(unsigned)((n + 255) / 256), 256, 0, stream>>>(p, n, v);
}

extern "C" void kernel_launch(void* const* d_in, const int* in_sizes, int n_in,
                              void* d_out, int out_size, void* d_ws, size_t ws_size,
                              hipStream_t stream) {
  // ---- inputs (setup_inputs dict order; params flattened in insertion order)
  const float* x        = (const float*)d_in[0];
  const int*   idx_k1   = (const int*)  d_in[1];
  const float* val_k1   = (const float*)d_in[2];
  const int*   idx_k2   = (const int*)  d_in[3];
  const float* val_k2   = (const float*)d_in[4];
  const int*   idx_k4   = (const int*)  d_in[5];
  const float* val_k4   = (const float*)d_in[6];
  const int*   api      = (const int*)  d_in[7];   // all_poly_idx [2,EA]
  const float* apv      = (const float*)d_in[8];   // all_poly_val [EA,10]
  const float* loop_val = (const float*)d_in[9];   // [N,10]
  const float* edge_attr= (const float*)d_in[10];  // [E1,H]
  const float* W_enc  = (const float*)d_in[11];
  const float* b_enc  = (const float*)d_in[12];
  const float* conv_W = (const float*)d_in[13];    // [NB,R,H,H]
  const float* conv_b = (const float*)d_in[14];    // [NB,R,H]
  const float* W_loop = (const float*)d_in[15];
  const float* b_loop = (const float*)d_in[16];
  const float* W_conn = (const float*)d_in[17];
  const float* b_conn = (const float*)d_in[18];
  const float* W_jk   = (const float*)d_in[19];
  const float* b_jk   = (const float*)d_in[20];
  const float* g_jk   = (const float*)d_in[21];
  const float* bb_jk  = (const float*)d_in[22];
  const float* Wq     = (const float*)d_in[23];
  const float* Wk     = (const float*)d_in[24];
  const float* Wv     = (const float*)d_in[25];
  const float* Wo     = (const float*)d_in[26];
  const float* we     = (const float*)d_in[27];
  const float* W_ffn1 = (const float*)d_in[28];
  const float* b_ffn1 = (const float*)d_in[29];
  const float* W_ffn2 = (const float*)d_in[30];
  const float* b_ffn2 = (const float*)d_in[31];
  const float* W_out  = (const float*)d_in[32];
  const float* b_out  = (const float*)d_in[33];

  // ---- workspace layout (floats)
  float* ws = (float*)d_ws;
  const ll NH = (ll)NNODE * HDIM;      // 12.8M
  float* h       = ws;                 // running node state
  float* nh0     = ws + 1  * NH;
  float* nh1     = ws + 2  * NH;
  float* nh2     = ws + 3  * NH;
  float* nh3     = ws + 4  * NH;
  float* nh4     = ws + 5  * NH;
  float* looph   = ws + 6  * NH;
  float* agg     = ws + 7  * NH;
  float* qb      = ws + 8  * NH;
  float* kb      = ws + 9  * NH;
  float* vb      = ws + 10 * NH;
  float* attnagg = ws + 11 * NH;
  float* yjk     = ws + 12 * NH;
  float* mid     = ws + 13 * NH;       // [N, 2H] -> 2*NH
  float* conn    = ws + 15 * NH;       // [EA, H] = 6*NH
  float* nhcat   = ws + 21 * NH;       // [N, 6H] = 6*NH
  float* meanb   = ws + 27 * NH;
  float* rstdb   = meanb + HDIM;
  float* mbuf    = rstdb + HDIM;       // [N]
  float* Zbuf    = mbuf + NNODE;       // [N]
  float* score   = Zbuf + NNODE;       // [EA]
  float* prb     = score + EAC;        // [EA]

  // 1) encoder: nh0 = x @ W_enc + b_enc
  launch_gemm(x, W_enc, b_enc, nullptr, nh0, NNODE, DIN, HDIM, 0, stream);
  hipMemcpyAsync(h, nh0, NH * sizeof(float), hipMemcpyDeviceToDevice, stream);

  // 2) conv blocks
  const int*   eidx[NB]  = { idx_k1, idx_k2, idx_k4, api };
  const float* eval_[NB] = { val_k1, val_k2, val_k4, apv + (EMB - 1) };
  const int    ecnt[NB]  = { E1C, E2C, E4C, EAC };
  const int    evst[NB]  = { 1, 1, 1, EMB };
  float* snaps[NB] = { nh1, nh2, nh3, nh4 };
  for (int l = 0; l < NB; ++l) {
    for (int r = 0; r < RREP; ++r) {
      launch_fill(agg, NH, 0.f, stream);
      ll work = (ll)ecnt[l] * HDIM;
      spmm_scatter<<<(unsigned)((work + 255) / 256), 256, 0, stream>>>(
          h, eidx[l], eval_[l], evst[l], agg, ecnt[l]);
      const float* W = conv_W + ((ll)(l * RREP + r)) * HDIM * HDIM;
      const float* b = conv_b + ((ll)(l * RREP + r)) * HDIM;
      launch_gemm(agg, W, b, h, h, NNODE, HDIM, HDIM, 1, stream);   // h += relu(agg@W+b)
    }
    hipMemcpyAsync(snaps[l], h, NH * sizeof(float), hipMemcpyDeviceToDevice, stream);
  }

  // 3) PE layers
  launch_gemm(loop_val, W_loop, b_loop, nullptr, looph, NNODE, EMB, HDIM, 0, stream);
  launch_gemm(apv,      W_conn, b_conn, nullptr, conn,  EAC,   EMB, HDIM, 0, stream);

  // 4) JK concat -> nhcat [N, 6H]
  {
    const float* parts[6] = { nh0, nh1, nh2, nh3, nh4, looph };
    unsigned blocks = (unsigned)((NH + 255) / 256);
    for (int p = 0; p < 6; ++p)
      copy_part<<<blocks, 256, 0, stream>>>(parts[p], nhcat, NNODE, 6 * HDIM, p * HDIM);
  }

  // 5) JK MLP: yjk = nhcat @ W_jk + b_jk ; then BN + ReLU -> h
  launch_gemm(nhcat, W_jk, b_jk, nullptr, yjk, NNODE, 6 * HDIM, HDIM, 0, stream);
  bn_stats<<<HDIM, 256, 0, stream>>>(yjk, meanb, rstdb, NNODE);
  bn_relu<<<(unsigned)((NH + 255) / 256), 256, 0, stream>>>(
      yjk, meanb, rstdb, g_jk, bb_jk, h, NH);

  // 6) attention
  launch_gemm(h, Wq, nullptr, nullptr, qb, NNODE, HDIM, HDIM, 0, stream);
  launch_gemm(h, Wk, nullptr, nullptr, kb, NNODE, HDIM, HDIM, 0, stream);
  launch_gemm(h, Wv, nullptr, nullptr, vb, NNODE, HDIM, HDIM, 0, stream);
  add_inplace<<<(unsigned)(((ll)E1C * HDIM + 255) / 256), 256, 0, stream>>>(
      conn, edge_attr, (ll)E1C * HDIM);               // full_val = conn[:E1] + edge_attr
  launch_fill(mbuf, NNODE, -__builtin_inff(), stream);
  launch_fill(Zbuf, NNODE, 0.f, stream);
  launch_fill(attnagg, NH, 0.f, stream);
  attn_score<<<(EAC + 7) / 8, 256, 0, stream>>>(qb, kb, conn, we, api, score, mbuf, EAC);
  attn_pr<<<(EAC + 255) / 256, 256, 0, stream>>>(score, mbuf, api, prb, Zbuf, EAC);
  attn_aggregate<<<(unsigned)(((ll)EAC * HDIM + 255) / 256), 256, 0, stream>>>(
      prb, Zbuf, vb, api, attnagg, EAC);
  launch_gemm(attnagg, Wo, nullptr, h, h, NNODE, HDIM, HDIM, 3, stream);  // h += agg@Wo

  // 7) FFN: h = h + relu(h@W1+b1)@W2 + b2
  launch_gemm(h,   W_ffn1, b_ffn1, nullptr, mid, NNODE, HDIM,     2 * HDIM, 2, stream);
  launch_gemm(mid, W_ffn2, b_ffn2, h,       h,   NNODE, 2 * HDIM, HDIM,     3, stream);

  // 8) head: out = h @ W_out + b_out   [N, 10]
  launch_gemm(h, W_out, b_out, nullptr, (float*)d_out, NNODE, HDIM, DOUT, 0, stream);
}
